// DetectionTarget_82231443849474
// MI455X (gfx1250) — matrix operand order repaired and checked by reading, not compiled, not run on hardware
//
#include <hip/hip_runtime.h>
#include <hip/hip_bf16.h>

// ---------------------------------------------------------------------------
// DetectionTargetLayer (Mask R-CNN target sampling) for MI455X / gfx1250.
//   Kernel 1 (dt_sample): per-image IoU, pos/neg sampling, deltas, packing.
//   Kernel 2 (dt_masks):  per-ROI crop_and_resize as a separable bilinear
//                         resample expressed as a 32x64 x 64x32 fp16 GEMM,
//                         executed with v_wmma_f32_16x16x32_f16 (8 per ROI).
// ---------------------------------------------------------------------------

typedef __attribute__((ext_vector_type(16))) _Float16 v16h;
typedef __attribute__((ext_vector_type(8)))  _Float16 v8h;
typedef __attribute__((ext_vector_type(8)))  float    v8f;

#define NPROP   2000
#define NGT     64
#define T_ROIS  200
#define P_CAP   66          // int(200 * 0.33)
#define MH      28
#define MW      28
#define HH      512
#define WW      512
#define GG      64

__device__ __forceinline__ v16h cat8(v8h lo, v8h hi) {
    return __builtin_shufflevector(lo, hi, 0, 1, 2, 3, 4, 5, 6, 7,
                                   8, 9, 10, 11, 12, 13, 14, 15);
}

// Deterministic counter-based uniform in [0,1): replaces jax.random streams
// (bit-exact replication of Threefry keys is not possible; this keeps the
// launch deterministic across graph replays as the harness requires).
__device__ __forceinline__ float rnd_uniform(unsigned b, unsigned n, unsigned salt) {
    unsigned x = (b * 0x9E3779B9u) ^ (n * 0x85EBCA6Bu) ^ (salt * 0xC2B2AE35u) ^ 0x68BC21EBu;
    x ^= x >> 16; x *= 0x7FEB352Du;
    x ^= x >> 15; x *= 0x846CA68Bu;
    x ^= x >> 16;
    return (float)(x >> 8) * (1.0f / 16777216.0f);
}

// ---------------------------------------------------------------------------
// Kernel 1: one 256-thread block per image (8 wave32s on one WGP).
// Priority arrays live in LDS (~21 KB of the 320 KB pool).
// ---------------------------------------------------------------------------
__global__ __launch_bounds__(256)
void dt_sample_kernel(const float* __restrict__ proposals,     // [B][NPROP][4]
                      const int*   __restrict__ gt_class_ids,  // [B][NGT]
                      const float* __restrict__ gt_boxes,      // [B][NGT][4]
                      float* __restrict__ out_rois,            // [B][T][4]
                      float* __restrict__ out_cls,             // [B][T]   (float-cast int)
                      float* __restrict__ out_deltas,          // [B][T][4]
                      float* __restrict__ ws_box,              // [B][P_CAP][4]
                      int*   __restrict__ ws_assign,           // [B][P_CAP]
                      int*   __restrict__ ws_npos)             // [B]
{
    const int b   = blockIdx.x;
    const int tid = threadIdx.x;

    __shared__ float         s_gt[NGT][4];
    __shared__ float         s_area[NGT];
    __shared__ int           s_cid[NGT];
    __shared__ unsigned char s_noncrowd[NGT];
    __shared__ unsigned char s_crowd[NGT];
    __shared__ float         s_pospri[NPROP];
    __shared__ float         s_negpri[NPROP];
    __shared__ float         s_rv[256];
    __shared__ int           s_ri[256];
    __shared__ int           s_pos_idx[P_CAP];
    __shared__ int           s_neg_idx[T_ROIS];
    __shared__ int           s_cnt[2];     // pos_count, neg_count

    const float* prop = proposals + (size_t)b * NPROP * 4;

    if (tid < NGT) {
        const float g0 = gt_boxes[((size_t)b * NGT + tid) * 4 + 0];
        const float g1 = gt_boxes[((size_t)b * NGT + tid) * 4 + 1];
        const float g2 = gt_boxes[((size_t)b * NGT + tid) * 4 + 2];
        const float g3 = gt_boxes[((size_t)b * NGT + tid) * 4 + 3];
        s_gt[tid][0] = g0; s_gt[tid][1] = g1; s_gt[tid][2] = g2; s_gt[tid][3] = g3;
        s_area[tid]  = (g2 - g0) * (g3 - g1);
        const int cid = gt_class_ids[(size_t)b * NGT + tid];
        s_cid[tid] = cid;
        const bool vg = (g0 != 0.0f) || (g1 != 0.0f) || (g2 != 0.0f) || (g3 != 0.0f);
        s_noncrowd[tid] = (unsigned char)(vg && cid > 0);
        s_crowd[tid]    = (unsigned char)(vg && cid < 0);
    }
    if (tid == 0) { s_cnt[0] = 0; s_cnt[1] = 0; }
    __syncthreads();

    // --- per-proposal IoU maxima, eligibility, random priorities -----------
    int local_pos = 0, local_neg = 0;
    for (int n = tid; n < NPROP; n += 256) {
        const float p0 = prop[n * 4 + 0];
        const float p1 = prop[n * 4 + 1];
        const float p2 = prop[n * 4 + 2];
        const float p3 = prop[n * 4 + 3];
        const bool vp = (p0 != 0.0f) || (p1 != 0.0f) || (p2 != 0.0f) || (p3 != 0.0f);
        const float pa = (p2 - p0) * (p3 - p1);
        float rmax = -1.0f;   // ov.max (ov == -1 where not non-crowd)
        float cmax = 0.0f;    // crowd IoU max
        for (int g = 0; g < NGT; ++g) {
            const float yy1 = fmaxf(p0, s_gt[g][0]);
            const float xx1 = fmaxf(p1, s_gt[g][1]);
            const float yy2 = fminf(p2, s_gt[g][2]);
            const float xx2 = fminf(p3, s_gt[g][3]);
            const float inter = fmaxf(yy2 - yy1, 0.0f) * fmaxf(xx2 - xx1, 0.0f);
            const float uni   = fmaxf(pa + s_area[g] - inter, 1e-12f);
            const float iou   = inter / uni;
            if (s_noncrowd[g]) rmax = fmaxf(rmax, iou);
            if (s_crowd[g])    cmax = fmaxf(cmax, iou);
        }
        const bool pos = vp && (rmax >= 0.5f);
        const bool neg = vp && (rmax < 0.5f) && (cmax < 0.001f);
        s_pospri[n] = pos ? rnd_uniform((unsigned)b, (unsigned)n, 1u) : -1.0f;
        s_negpri[n] = neg ? rnd_uniform((unsigned)b, (unsigned)n, 2u) : -1.0f;
        local_pos += pos ? 1 : 0;
        local_neg += neg ? 1 : 0;
    }
    atomicAdd(&s_cnt[0], local_pos);
    atomicAdd(&s_cnt[1], local_neg);
    __syncthreads();

    const int pos_count = s_cnt[0];
    const int neg_count = s_cnt[1];
    const int n_pos = min(pos_count, P_CAP);
    int neg_needed = (n_pos > 0) ? (int)((float)n_pos / 0.33f) - n_pos : T_ROIS;
    int n_neg = min(min(neg_needed, neg_count), T_ROIS - n_pos);
    if (n_neg < 0) n_neg = 0;

    // --- iterative shared-memory argmax == top_k over random priorities ----
    for (int k = 0; k < n_pos; ++k) {
        float best = -2.0f; int bi = 0;
        for (int n = tid; n < NPROP; n += 256) {
            const float v = s_pospri[n];
            if (v > best) { best = v; bi = n; }
        }
        s_rv[tid] = best; s_ri[tid] = bi;
        __syncthreads();
        for (int st = 128; st > 0; st >>= 1) {
            if (tid < st && s_rv[tid + st] > s_rv[tid]) {
                s_rv[tid] = s_rv[tid + st]; s_ri[tid] = s_ri[tid + st];
            }
            __syncthreads();
        }
        if (tid == 0) { s_pos_idx[k] = s_ri[0]; s_pospri[s_ri[0]] = -3.0f; }
        __syncthreads();
    }
    for (int k = 0; k < n_neg; ++k) {
        float best = -2.0f; int bi = 0;
        for (int n = tid; n < NPROP; n += 256) {
            const float v = s_negpri[n];
            if (v > best) { best = v; bi = n; }
        }
        s_rv[tid] = best; s_ri[tid] = bi;
        __syncthreads();
        for (int st = 128; st > 0; st >>= 1) {
            if (tid < st && s_rv[tid + st] > s_rv[tid]) {
                s_rv[tid] = s_rv[tid + st]; s_ri[tid] = s_ri[tid + st];
            }
            __syncthreads();
        }
        if (tid == 0) { s_neg_idx[k] = s_ri[0]; s_negpri[s_ri[0]] = -3.0f; }
        __syncthreads();
    }

    // --- pack [positives | negatives | zeros] into the T fixed slots -------
    for (int i = tid; i < T_ROIS; i += 256) {
        float r0 = 0.f, r1 = 0.f, r2 = 0.f, r3 = 0.f;
        float cls = 0.f;
        float d0 = 0.f, d1 = 0.f, d2 = 0.f, d3 = 0.f;
        if (i < n_pos) {
            const int idx = s_pos_idx[i];
            const float p0 = prop[idx * 4 + 0];
            const float p1 = prop[idx * 4 + 1];
            const float p2 = prop[idx * 4 + 2];
            const float p3 = prop[idx * 4 + 3];
            const float pa = (p2 - p0) * (p3 - p1);
            // assignment = argmax over non-crowd gt of IoU (first-max like argmax)
            float best = -1.0f; int bg = 0;
            for (int g = 0; g < NGT; ++g) {
                float v = -1.0f;
                if (s_noncrowd[g]) {
                    const float yy1 = fmaxf(p0, s_gt[g][0]);
                    const float xx1 = fmaxf(p1, s_gt[g][1]);
                    const float yy2 = fminf(p2, s_gt[g][2]);
                    const float xx2 = fminf(p3, s_gt[g][3]);
                    const float inter = fmaxf(yy2 - yy1, 0.0f) * fmaxf(xx2 - xx1, 0.0f);
                    const float uni   = fmaxf(pa + s_area[g] - inter, 1e-12f);
                    v = inter / uni;
                }
                if (v > best) { best = v; bg = g; }
            }
            const float g0 = s_gt[bg][0], g1 = s_gt[bg][1];
            const float g2 = s_gt[bg][2], g3 = s_gt[bg][3];
            cls = (float)s_cid[bg];
            const float h  = p2 - p0, w  = p3 - p1;
            const float cy = p0 + 0.5f * h, cx = p1 + 0.5f * w;
            const float gh = g2 - g0, gw = g3 - g1;
            const float gcy = g0 + 0.5f * gh, gcx = g1 + 0.5f * gw;
            d0 = ((gcy - cy) / h) / 0.1f;
            d1 = ((gcx - cx) / w) / 0.1f;
            d2 = logf(gh / h) / 0.2f;
            d3 = logf(gw / w) / 0.2f;
            r0 = p0; r1 = p1; r2 = p2; r3 = p3;
            ws_box[((size_t)b * P_CAP + i) * 4 + 0] = p0;
            ws_box[((size_t)b * P_CAP + i) * 4 + 1] = p1;
            ws_box[((size_t)b * P_CAP + i) * 4 + 2] = p2;
            ws_box[((size_t)b * P_CAP + i) * 4 + 3] = p3;
            ws_assign[(size_t)b * P_CAP + i] = bg;
        } else if (i < n_pos + n_neg) {
            const int idx = s_neg_idx[i - n_pos];
            r0 = prop[idx * 4 + 0];
            r1 = prop[idx * 4 + 1];
            r2 = prop[idx * 4 + 2];
            r3 = prop[idx * 4 + 3];
        }
        const size_t o = ((size_t)b * T_ROIS + i);
        out_rois[o * 4 + 0] = r0; out_rois[o * 4 + 1] = r1;
        out_rois[o * 4 + 2] = r2; out_rois[o * 4 + 3] = r3;
        out_cls[o] = cls;
        out_deltas[o * 4 + 0] = d0; out_deltas[o * 4 + 1] = d1;
        out_deltas[o * 4 + 2] = d2; out_deltas[o * 4 + 3] = d3;
    }
    if (tid == 0) ws_npos[b] = n_pos;
}

// ---------------------------------------------------------------------------
// Kernel 2: one wave32 per ROI slot. Separable bilinear crop_and_resize:
//   A [32x64] (LDS, f16)  = row-interpolated mask samples at the <=56 needed
//                           x columns (K compacted to 2 per output column),
//   Bt[32x64] (LDS, f16)  = TRANSPOSED x-weight matrix: row n holds its two
//                           non-zeros at k=2n, 2n+1  -> per-lane fragment is
//                           two 16-byte runs 32 bytes apart (two ds_load_b128),
//   out[32x32] = A x B via 4 accumulator tiles x 2 K-steps of
//                v_wmma_f32_16x16x32_f16 (fully unrolled: 8 straight-line).
// Results staged through a 32x32 f32 LDS tile for coalesced global stores.
// Fragment packing per CDNA5 ISA 7.12.2 (wave32 layouts).
// ---------------------------------------------------------------------------
__global__ __launch_bounds__(32)
void dt_masks_kernel(const unsigned char* __restrict__ gt_masks, // [B][512][512][64]
                     const float* __restrict__ ws_box,           // [B][P_CAP][4]
                     const int*   __restrict__ ws_assign,        // [B][P_CAP]
                     const int*   __restrict__ ws_npos,          // [B]
                     float* __restrict__ out_masks)              // [B][T][28][28]
{
    const int bs   = blockIdx.x;           // b * T_ROIS + s
    const int b    = bs / T_ROIS;
    const int s    = bs % T_ROIS;
    const int lane = threadIdx.x;
    float* outp = out_masks + (size_t)bs * (MH * MW);

    const int npos = ws_npos[b];
    if (s >= npos) {                       // uniform per block: zero-fill slot
        for (int t = lane; t < MH * MW; t += 32) outp[t] = 0.0f;
        return;
    }

    __shared__ alignas(16) _Float16 As[32][64];   // rows 28..31, cols 56..63 stay 0
    __shared__ alignas(16) _Float16 Bts[32][64];  // rows 28..31, cols 56..63 stay 0
    __shared__ alignas(16) float    Cs[32][32];   // D staging for coalesced stores
    {
        int* za = (int*)&As[0][0];
        int* zb = (int*)&Bts[0][0];
        for (int t = lane; t < 1024; t += 32) { za[t] = 0; zb[t] = 0; }
    }

    const float y1 = ws_box[((size_t)b * P_CAP + s) * 4 + 0];
    const float x1 = ws_box[((size_t)b * P_CAP + s) * 4 + 1];
    const float y2 = ws_box[((size_t)b * P_CAP + s) * 4 + 2];
    const float x2 = ws_box[((size_t)b * P_CAP + s) * 4 + 3];
    const int   g  = ws_assign[(size_t)b * P_CAP + s];
    const unsigned char* M = gt_masks + (size_t)b * HH * WW * GG + g;

    const float sy = (y2 - y1) * (float)(HH - 1) / (float)(MH - 1);
    const float sx = (x2 - x1) * (float)(WW - 1) / (float)(MW - 1);
    const float oy = y1 * (float)(HH - 1);
    const float ox = x1 * (float)(WW - 1);

    // Warm L2 lines for the strided byte gathers (gfx1250 global_prefetch_b8).
    if (lane < MH) {
        const float ys = oy + (float)lane * sy;
        const float y0 = floorf(ys);
        const int y0i = (int)fminf(fmaxf(y0, 0.0f), (float)(HH - 1));
        const int y1i = (int)fminf(fmaxf(y0 + 1.0f, 0.0f), (float)(HH - 1));
        const int xi  = (int)fminf(fmaxf(floorf(ox), 0.0f), (float)(WW - 1));
        __builtin_prefetch(M + ((size_t)y0i * WW + xi) * GG, 0, 0);
        __builtin_prefetch(M + ((size_t)y1i * WW + xi) * GG, 0, 0);
    }

    // Bt matrix (transposed weights): row n holds (1-wx), wx at k = 2n, 2n+1.
    if (lane < MW) {
        const float xs = ox + (float)lane * sx;
        const float x0 = floorf(xs);
        const float wx = xs - x0;
        const bool xin = (xs >= 0.0f) && (xs <= (float)(WW - 1));
        Bts[lane][2 * lane + 0] = (_Float16)(xin ? (1.0f - wx) : 0.0f);
        Bts[lane][2 * lane + 1] = (_Float16)(xin ? wx : 0.0f);
    }

    // A matrix: A[i][2j+t] = y-interpolated sample of column x_t(j) in row i.
    for (int t = lane; t < MH * 2 * MW; t += 32) {
        const int i   = t / (2 * MW);
        const int k   = t % (2 * MW);
        const int j   = k >> 1;
        const int sel = k & 1;
        const float ys = oy + (float)i * sy;
        const float y0 = floorf(ys);
        const float wy = ys - y0;
        const bool yin = (ys >= 0.0f) && (ys <= (float)(HH - 1));
        const float xs = ox + (float)j * sx;
        const int xi  = (int)fminf(fmaxf(floorf(xs) + (float)sel, 0.0f), (float)(WW - 1));
        const int y0i = (int)fminf(fmaxf(y0, 0.0f), (float)(HH - 1));
        const int y1i = (int)fminf(fmaxf(y0 + 1.0f, 0.0f), (float)(HH - 1));
        const float m0 = (float)M[((size_t)y0i * WW + xi) * GG];
        const float m1 = (float)M[((size_t)y1i * WW + xi) * GG];
        const float v = yin ? ((1.0f - wy) * m0 + wy * m1) : 0.0f;
        As[i][k] = (_Float16)v;
    }
    __syncthreads();

    // GEMM 32x64 x 64x32 via v_wmma_f32_16x16x32_f16 (EXEC all-ones here).
    // A fragment: lane holds row M=lane&15; halves 0..7 = K rowk8+0..7,
    //             halves 8..15 = K 16+rowk8+0..7  -> two ds_load_b128.
    // B fragment: lane holds col N=lane&15; halves 0..7 = K koff+0..7,
    //             halves 8..15 = K 16+koff+0..7 of Bt row N -> two ds_load_b128.
    v8f acc[2][2] = {};
    const int arow  = lane & 15;
    const int rowk8 = (lane & 16) ? 8 : 0;
    const int koff  = (lane & 16) ? 8 : 0;
#pragma unroll
    for (int kb = 0; kb < 2; ++kb) {
        v16h af[2], bf[2];
#pragma unroll
        for (int mi = 0; mi < 2; ++mi) {
            const _Float16* ap = &As[mi * 16 + arow][kb * 32 + rowk8];
            af[mi] = cat8(*(const v8h*)ap, *(const v8h*)(ap + 16));
        }
#pragma unroll
        for (int ni = 0; ni < 2; ++ni) {
            const _Float16* bp = &Bts[ni * 16 + arow][kb * 32 + koff];
            bf[ni] = cat8(*(const v8h*)bp, *(const v8h*)(bp + 16));
        }
#pragma unroll
        for (int mi = 0; mi < 2; ++mi)
#pragma unroll
            for (int ni = 0; ni < 2; ++ni)
                acc[mi][ni] = __builtin_amdgcn_wmma_f32_16x16x32_f16(
                    false, af[mi], false, bf[ni],
                    (short)0, acc[mi][ni], false, false);
    }

    // D layout (32-bit C/D 16x16): N = lane&15, M = vgpr + 8*(lane>=16).
    // Stage full 32x32 tile in LDS, then linear coalesced copy with
    // round-half-to-even (jnp.round semantics).
    const int mhalf = (lane >> 4) * 8;
#pragma unroll
    for (int mi = 0; mi < 2; ++mi)
#pragma unroll
        for (int ni = 0; ni < 2; ++ni)
#pragma unroll
            for (int r = 0; r < 8; ++r)
                Cs[mi * 16 + mhalf + r][ni * 16 + arow] = acc[mi][ni][r];
    __syncthreads();
    for (int t = lane; t < MH * MW; t += 32)
        outp[t] = rintf(Cs[t / MW][t % MW]);
}

// ---------------------------------------------------------------------------
extern "C" void kernel_launch(void* const* d_in, const int* in_sizes, int n_in,
                              void* d_out, int out_size, void* d_ws, size_t ws_size,
                              hipStream_t stream) {
    const float*         proposals = (const float*)d_in[0];
    const int*           gt_cid    = (const int*)d_in[1];
    const float*         gt_boxes  = (const float*)d_in[2];
    const unsigned char* gt_masks  = (const unsigned char*)d_in[3];

    const int B = in_sizes[0] / (NPROP * 4);

    float* out        = (float*)d_out;
    float* out_rois   = out;                                   // [B][T][4]
    float* out_cls    = out_rois + (size_t)B * T_ROIS * 4;     // [B][T]
    float* out_deltas = out_cls + (size_t)B * T_ROIS;          // [B][T][4]
    float* out_masks  = out_deltas + (size_t)B * T_ROIS * 4;   // [B][T][28][28]

    float* ws_box    = (float*)d_ws;                           // [B][P_CAP][4]
    int*   ws_assign = (int*)(ws_box + (size_t)B * P_CAP * 4); // [B][P_CAP]
    int*   ws_npos   = ws_assign + (size_t)B * P_CAP;          // [B]

    dt_sample_kernel<<<B, 256, 0, stream>>>(proposals, gt_cid, gt_boxes,
                                            out_rois, out_cls, out_deltas,
                                            ws_box, ws_assign, ws_npos);
    dt_masks_kernel<<<B * T_ROIS, 32, 0, stream>>>(gt_masks, ws_box, ws_assign,
                                                   ws_npos, out_masks);
}
